// QuantizedLinear_2654289789344
// MI455X (gfx1250) — compile-verified, matrix operand order checked by reading, not run on hardware
//
#include <hip/hip_runtime.h>

// Native (ext_vector) types so WMMA builtins and nontemporal loads work.
typedef __attribute__((ext_vector_type(16))) _Float16 v16h;
typedef __attribute__((ext_vector_type(8)))  _Float16 v8h;
typedef __attribute__((ext_vector_type(4)))  _Float16 v4h;
typedef __attribute__((ext_vector_type(8)))  float    v8f;
typedef __attribute__((ext_vector_type(4)))  float    f4;
typedef __attribute__((ext_vector_type(4)))  int      i4;

#define K_DIM   4096
#define N_DIM   11008
#define M_DIM   64
#define K_SPLIT 4
#define K_CHUNK (K_DIM / K_SPLIT)   // 1024 -> 32 WMMA K-steps per wave

// ---------------------------------------------------------------------------
// Kernel 1: convert x (f32) -> f16 staging buffer in workspace (512 KB).
// ---------------------------------------------------------------------------
__global__ void __launch_bounds__(256)
xcvt_f32_to_f16(const float* __restrict__ x, _Float16* __restrict__ xh, int n4) {
  int i = blockIdx.x * blockDim.x + threadIdx.x;
  if (i < n4) {
    f4 v = *(const f4*)(x + (size_t)i * 4);
    v4h h;
    h.x = (_Float16)v.x; h.y = (_Float16)v.y;
    h.z = (_Float16)v.z; h.w = (_Float16)v.w;
    *(v4h*)(xh + (size_t)i * 4) = h;
  }
}

// ---------------------------------------------------------------------------
// Kernel 2: seed out[m, o] = bias[o] so K-split partials can atomically merge.
// grid = (N_DIM/256, M_DIM), coalesced b32 stores.
// ---------------------------------------------------------------------------
__global__ void __launch_bounds__(256)
bias_init(const float* __restrict__ bias, float* __restrict__ out) {
  const int o = blockIdx.x * 256 + threadIdx.x;
  const int m = blockIdx.y;
  out[(size_t)m * N_DIM + o] = bias[o];
}

// ---------------------------------------------------------------------------
// Kernel 3: WMMA GEMM, K-split x4 for memory-level parallelism (2752 waves).
// One wave: 16-wide N-tile x all 64 M rows x 1024-deep K chunk. Each int32
// weight is loaded from HBM exactly once (non-temporal) and reused 4x from
// registers. The 4 K-chunk waves of one N-tile share a block; each
// reads A from its own 1024-wide K chunk of x.
//
// Layouts (CDNA5 ISA 7.12.2, wave32):
//   B (32x16 f16): lane = N column (lane&15); lane>=16 handles K offset +8.
//     v16h elems 0..7 = K k0..k0+7, elems 8..15 = K k0+16..k0+23.
//   A (16x32 f16): lane = M row (lane&15); same K striping.
//   C/D (16x16 f32): VGPR r -> M = r + (lane>=16 ? 8 : 0), N = lane&15.
// ---------------------------------------------------------------------------
__global__ void __launch_bounds__(128)
qlinear_wmma_f16(const _Float16* __restrict__ xh,
                 const int*      __restrict__ w,
                 const float*    __restrict__ scale,
                 float*          __restrict__ out) {
  const int lane = threadIdx.x & 31;
  const int wave = threadIdx.x >> 5;           // 4 waves / block
  const int ln   = lane & 15;                  // N column within tile (B/C/D)
  const int hi   = lane >> 4;                  // lane-group: K/M offset select
  const int gw   = blockIdx.x * 4 + wave;      // global wave id
  const int ntile = gw >> 2;                   // [0, 688)
  const int kc    = gw & 3;                    // K chunk [0, 4)
  const int o     = ntile * 16 + ln;           // output feature

  const int*      __restrict__ wrow  = w  + (size_t)o * K_DIM + (size_t)kc * K_CHUNK;
  const _Float16* __restrict__ xbase = xh + (size_t)kc * K_CHUNK;

  v8f acc0 = {}; v8f acc1 = {}; v8f acc2 = {}; v8f acc3 = {};

  const int khi = hi * 8;

  for (int kb = 0; kb < K_CHUNK; kb += 32) {
    const int k0 = kb + khi;

    // ---- B tile: 32 int8-in-int32 weights, single-use -> non-temporal ----
    i4 w0 = __builtin_nontemporal_load((const i4*)(wrow + k0));
    i4 w1 = __builtin_nontemporal_load((const i4*)(wrow + k0 + 4));
    i4 w2 = __builtin_nontemporal_load((const i4*)(wrow + k0 + 16));
    i4 w3 = __builtin_nontemporal_load((const i4*)(wrow + k0 + 20));

    v16h b;
    b[0]  = (_Float16)w0.x; b[1]  = (_Float16)w0.y;
    b[2]  = (_Float16)w0.z; b[3]  = (_Float16)w0.w;
    b[4]  = (_Float16)w1.x; b[5]  = (_Float16)w1.y;
    b[6]  = (_Float16)w1.z; b[7]  = (_Float16)w1.w;
    b[8]  = (_Float16)w2.x; b[9]  = (_Float16)w2.y;
    b[10] = (_Float16)w2.z; b[11] = (_Float16)w2.w;
    b[12] = (_Float16)w3.x; b[13] = (_Float16)w3.y;
    b[14] = (_Float16)w3.z; b[15] = (_Float16)w3.w;

    // ---- 4 M-tiles reuse the B register tile ----
#define MTILE(T, ACC)                                                          \
    {                                                                          \
      const _Float16* xr = xbase + (size_t)((T) * 16 + ln) * K_DIM + k0;       \
      v8h alo = *(const v8h*)(xr);                                             \
      v8h ahi = *(const v8h*)(xr + 16);                                        \
      v16h a = __builtin_shufflevector(alo, ahi, 0, 1, 2, 3, 4, 5, 6, 7,       \
                                       8, 9, 10, 11, 12, 13, 14, 15);          \
      ACC = __builtin_amdgcn_wmma_f32_16x16x32_f16(                            \
          /*neg_a=*/false, a, /*neg_b=*/false, b,                              \
          /*c_mod=*/(short)0, ACC, /*reuse_a=*/false, /*reuse_b=*/false);      \
    }
    MTILE(0, acc0)
    MTILE(1, acc1)
    MTILE(2, acc2)
    MTILE(3, acc3)
#undef MTILE
  }

  // ---- Epilogue: dequant scale, merge K-split partials with f32 atomics ----
  const float s = scale[o] * (1.0f / 127.0f);

#define STORE_TILE(T, ACC)                                                     \
  {                                                                            \
    float* orow = out + (size_t)((T) * 16 + hi * 8) * N_DIM + o;               \
    _Pragma("unroll")                                                          \
    for (int r = 0; r < 8; ++r) {                                              \
      (void)__hip_atomic_fetch_add(&orow[(size_t)r * N_DIM], ACC[r] * s,       \
                                   __ATOMIC_RELAXED,                           \
                                   __HIP_MEMORY_SCOPE_AGENT);                  \
    }                                                                          \
  }
  STORE_TILE(0, acc0)
  STORE_TILE(1, acc1)
  STORE_TILE(2, acc2)
  STORE_TILE(3, acc3)
#undef STORE_TILE
}

// ---------------------------------------------------------------------------
extern "C" void kernel_launch(void* const* d_in, const int* in_sizes, int n_in,
                              void* d_out, int out_size, void* d_ws, size_t ws_size,
                              hipStream_t stream) {
  const float* x     = (const float*)d_in[0];   // [64, 4096] f32
  const int*   wq    = (const int*)d_in[1];     // [11008, 4096] int8-in-int32
  const float* scale = (const float*)d_in[2];   // [11008]
  const float* bias  = (const float*)d_in[3];   // [11008]
  float*       out   = (float*)d_out;           // [64, 11008] f32
  _Float16*    xh    = (_Float16*)d_ws;         // 512 KB staging

  const int n_x = in_sizes[0];                  // 262144
  const int n4  = n_x / 4;
  xcvt_f32_to_f16<<<(n4 + 255) / 256, 256, 0, stream>>>(x, xh, n4);

  // Seed output with bias so the K-split atomic merge is bias-correct.
  bias_init<<<dim3(N_DIM / 256, M_DIM), 256, 0, stream>>>(bias, out);

  // 688 N-tiles x 4 K-chunks, 4 waves per 128-thread block -> 688 blocks.
  qlinear_wmma_f16<<<(N_DIM / 16) * K_SPLIT / 4, 128, 0, stream>>>(xh, wq, scale, out);
}
